// GCLayer_57655640981900
// MI455X (gfx1250) — compile-verified, matrix operand order checked by reading, not run on hardware
//
#include <hip/hip_runtime.h>
#include <hip/hip_bf16.h>

typedef __bf16 bf16_t;
typedef __attribute__((ext_vector_type(16))) __bf16 v16bf;
typedef __attribute__((ext_vector_type(8)))  __bf16 v8bf;
typedef __attribute__((ext_vector_type(8)))  float  v8f;

#define DFEAT 64
#define NORMF 100.0f
#define LNEPS 1e-5f

__device__ __forceinline__ float sigmoidf_(float v) { return 1.0f / (1.0f + __expf(-v)); }
__device__ __forceinline__ float siluf_(float v)    { return v * sigmoidf_(v); }

// ---------------------------------------------------------------------------
// Fragment layouts (wave32 WMMA 16x16x32 bf16):
//   A-frag element e (lane: m=l&15, half=l>>4): K = 32c + 8*half + (e<8 ? e : e+8)
//   B-frag (lane: n=l&15) mirrors the K map; value = W[n][K]  (B[k][n] = W[n][k])
//   C/D layout: vgpr r, lane l -> (M = r + 8*(l>>4), N = l&15)
//
// wfrag: [24][32 lanes][16] bf16  (pre-converted B fragments, 24 KB)
//   f = 0..7  : W_lin  tile t=f>>1, chunk c=f&1
//   f = 8..15 : W1a    (att_W1 cols 0..63)
//   f = 16..23: W1b    (att_W1 cols 64..127)
// sctab: [4][32][4] float : per (t, lane) {b1[n], W1[n][128], w2[n], 0}, n=16t+(l&15)
// ---------------------------------------------------------------------------
__global__ void pack_kernel(const float* __restrict__ Wl,   // [64][64]
                            const float* __restrict__ W1,   // [64][129]
                            const float* __restrict__ b1,   // [64]
                            const float* __restrict__ w2,   // [64]
                            bf16_t* __restrict__ wfrag,
                            float*  __restrict__ sctab)
{
    int lane = (int)(threadIdx.x & 31);
    int wv   = (int)(threadIdx.x >> 5);   // 0..7
    int half = lane >> 4;
    int m    = lane & 15;

    for (int f = wv; f < 24; f += 8) {
        int grp = f >> 3;                 // 0=W_lin, 1=W1a, 2=W1b
        int t   = (f & 7) >> 1;
        int c   = f & 1;
        int n   = 16 * t + m;
        const float* src;
        if (grp == 0)      src = Wl + (size_t)n * DFEAT + 32 * c + 8 * half;
        else if (grp == 1) src = W1 + (size_t)n * 129 + 32 * c + 8 * half;
        else               src = W1 + (size_t)n * 129 + 64 + 32 * c + 8 * half;
        bf16_t* dst = wfrag + ((size_t)f * 32 + lane) * 16;
#pragma unroll
        for (int e = 0; e < 8; ++e) { dst[e] = (bf16_t)src[e]; dst[8 + e] = (bf16_t)src[16 + e]; }
    }
    if (wv == 0) {
#pragma unroll
        for (int t = 0; t < 4; ++t) {
            int n = 16 * t + m;
            float* d = sctab + ((size_t)t * 32 + lane) * 4;
            d[0] = b1[n];
            d[1] = W1[(size_t)n * 129 + 128];   // dist column
            d[2] = w2[n];
            d[3] = 0.0f;
        }
    }
}

// ---------------------------------------------------------------------------
// Kernel A: x = h @ W_lin^T + b_lin ; also emit bf16 mirror xb for WMMA gathers.
// One wave per 16-node tile. 8x v_wmma_f32_16x16x32_bf16 per wave.
// ---------------------------------------------------------------------------
__global__ void proj_kernel(const float* __restrict__ h,
                            const bf16_t* __restrict__ wfrag,
                            const float* __restrict__ bl,   // [64]
                            float*  __restrict__ x,
                            bf16_t* __restrict__ xb,
                            int N)
{
    int wave = (int)((blockIdx.x * blockDim.x + threadIdx.x) >> 5);
    int lane = (int)(threadIdx.x & 31);
    int ntiles = N >> 4;
    if (wave >= ntiles) return;
    int r0   = wave << 4;
    int half = lane >> 4;
    int m    = lane & 15;
    const v16bf* wf = (const v16bf*)wfrag;

    v16bf a[2];
#pragma unroll
    for (int c = 0; c < 2; ++c) {
        const float* p = h + (size_t)(r0 + m) * DFEAT + 32 * c + 8 * half;
#pragma unroll
        for (int e = 0; e < 8; ++e) { a[c][e] = (bf16_t)p[e]; a[c][8 + e] = (bf16_t)p[16 + e]; }
    }

#pragma unroll
    for (int t = 0; t < 4; ++t) {
        int n = 16 * t + m;
        float bias = bl[n];
        v8f acc;
#pragma unroll
        for (int r = 0; r < 8; ++r) acc[r] = bias;
#pragma unroll
        for (int c = 0; c < 2; ++c) {
            v16bf b = wf[(size_t)(2 * t + c) * 32 + lane];
            acc = __builtin_amdgcn_wmma_f32_16x16x32_bf16(false, a[c], false, b,
                                                          (short)0, acc, false, false);
        }
#pragma unroll
        for (int r = 0; r < 8; ++r) {
            int gm = r0 + r + 8 * half;
            x [(size_t)gm * DFEAT + n] = acc[r];
            xb[(size_t)gm * DFEAT + n] = (bf16_t)acc[r];
        }
    }
}

// ---------------------------------------------------------------------------
// Kernel B: edge MLP + attention + atomic scatter.
// hdn = silu(x_row@W1a^T + x_col@W1b^T + dist*w1c + b1)    (4 col-tiles x 4 WMMA)
// att = sigmoid(hdn . w2 + b2) * edge_mask
// agg[row] += x[col] * att   (fp32 global atomics, L2-resident accumulator)
// One wave per 16-edge tile; xb gathers are L2 hits (38 MB << 192 MB L2).
// ---------------------------------------------------------------------------
__global__ void edge_kernel(const bf16_t* __restrict__ xb,
                            const float*  __restrict__ x,
                            const long long* __restrict__ edges, // [2][E] int64
                            const float* __restrict__ dist,      // [E]
                            const float* __restrict__ emask,     // [E]
                            const bf16_t* __restrict__ wfrag,
                            const float*  __restrict__ sctab,
                            const float*  __restrict__ b2,       // [1]
                            float* __restrict__ agg,             // [N][64]
                            int E)
{
    int wave = (int)((blockIdx.x * blockDim.x + threadIdx.x) >> 5);
    int lane = (int)(threadIdx.x & 31);
    int ntiles = E >> 4;
    if (wave >= ntiles) return;
    int tb   = wave << 4;
    int half = lane >> 4;
    int m    = lane & 15;
    const v16bf*  wf = (const v16bf*)wfrag;
    const float4* sc4 = (const float4*)sctab;

    int   nrow = (int)edges[tb + m];
    int   ncol = (int)edges[(size_t)E + tb + m];
    float dv   = dist[tb + m];
    float evm  = emask[tb + m];

    // Gather A fragments (bf16, two contiguous 16B loads per chunk per lane).
    v16bf ar[2], ac[2];
    {
        const bf16_t* pr = xb + (size_t)nrow * DFEAT;
        const bf16_t* pc = xb + (size_t)ncol * DFEAT;
#pragma unroll
        for (int c = 0; c < 2; ++c) {
            int k0 = 32 * c + 8 * half;
            v8bf r0 = *(const v8bf*)(pr + k0);
            v8bf r1 = *(const v8bf*)(pr + k0 + 16);
            v8bf c0 = *(const v8bf*)(pc + k0);
            v8bf c1 = *(const v8bf*)(pc + k0 + 16);
#pragma unroll
            for (int e = 0; e < 8; ++e) {
                ar[c][e] = r0[e]; ar[c][8 + e] = r1[e];
                ac[c][e] = c0[e]; ac[c][8 + e] = c1[e];
            }
        }
    }

    // dist of edge m=r+8*half, broadcast once (tile-invariant across t).
    float dm[8];
#pragma unroll
    for (int r = 0; r < 8; ++r) dm[r] = __shfl(dv, r + 8 * half, 32);

    float b2v = b2[0];
    float satt[8];
#pragma unroll
    for (int r = 0; r < 8; ++r) satt[r] = 0.0f;

#pragma unroll
    for (int t = 0; t < 4; ++t) {
        float4 sc = sc4[(size_t)t * 32 + lane];   // {b1, w1c, w2, 0}
        v8f acc;
#pragma unroll
        for (int r = 0; r < 8; ++r) acc[r] = sc.x + dm[r] * sc.y;
#pragma unroll
        for (int c = 0; c < 2; ++c) {             // x_row @ W1a^T
            v16bf b = wf[(size_t)(8 + 2 * t + c) * 32 + lane];
            acc = __builtin_amdgcn_wmma_f32_16x16x32_bf16(false, ar[c], false, b,
                                                          (short)0, acc, false, false);
        }
#pragma unroll
        for (int c = 0; c < 2; ++c) {             // x_col @ W1b^T
            v16bf b = wf[(size_t)(16 + 2 * t + c) * 32 + lane];
            acc = __builtin_amdgcn_wmma_f32_16x16x32_bf16(false, ac[c], false, b,
                                                          (short)0, acc, false, false);
        }
#pragma unroll
        for (int r = 0; r < 8; ++r) satt[r] += siluf_(acc[r]) * sc.z;
    }

    // Reduce over N (16 lanes per half) -> att per edge, broadcast to all lanes.
    float attv[16];
#pragma unroll
    for (int r = 0; r < 8; ++r) {
        float s = satt[r];
#pragma unroll
        for (int o = 1; o < 16; o <<= 1) s += __shfl_xor(s, o, 32);
        float em = __shfl(evm, r + 8 * half, 32);
        float a  = sigmoidf_(s + b2v) * em;
        attv[r]     = __shfl(a, 0, 32);   // m = r     (from half 0)
        attv[r + 8] = __shfl(a, 16, 32);  // m = r + 8 (from half 1)
    }

    // Scatter: agg[row_m][:] += x[col_m][:] * att_m  (lane covers cols l, l+32)
#pragma unroll
    for (int mm = 0; mm < 16; ++mm) {
        int rn = __shfl(nrow, mm, 32);
        int cn = __shfl(ncol, mm, 32);
        float a = attv[mm];
        const float* xp = x   + (size_t)cn * DFEAT;
        float*       op = agg + (size_t)rn * DFEAT;
        atomicAdd(op + lane,      xp[lane]      * a);
        atomicAdd(op + lane + 32, xp[lane + 32] * a);
    }
}

// ---------------------------------------------------------------------------
// Kernel C: out = silu(LayerNorm(agg/100 + x) * g + b), in place over d_out.
// One wave per node; lane holds cols {l, l+32}; wave shfl_xor reductions.
// ---------------------------------------------------------------------------
__global__ void finalize_kernel(float* __restrict__ out,
                                const float* __restrict__ x,
                                const float* __restrict__ g,
                                const float* __restrict__ bta,
                                int N)
{
    int wave = (int)((blockIdx.x * blockDim.x + threadIdx.x) >> 5);
    int lane = (int)(threadIdx.x & 31);
    if (wave >= N) return;
    size_t base = (size_t)wave * DFEAT;

    float o0 = out[base + lane]      * (1.0f / NORMF) + x[base + lane];
    float o1 = out[base + lane + 32] * (1.0f / NORMF) + x[base + lane + 32];

    float s = o0 + o1;
#pragma unroll
    for (int o = 1; o < 32; o <<= 1) s += __shfl_xor(s, o, 32);
    float mu = s * (1.0f / 64.0f);

    float d0 = o0 - mu, d1 = o1 - mu;
    float v = d0 * d0 + d1 * d1;
#pragma unroll
    for (int o = 1; o < 32; o <<= 1) v += __shfl_xor(v, o, 32);
    float inv = rsqrtf(v * (1.0f / 64.0f) + LNEPS);

    float y0 = d0 * inv * g[lane]      + bta[lane];
    float y1 = d1 * inv * g[lane + 32] + bta[lane + 32];
    out[base + lane]      = siluf_(y0);
    out[base + lane + 32] = siluf_(y1);
}

extern "C" void kernel_launch(void* const* d_in, const int* in_sizes, int n_in,
                              void* d_out, int out_size, void* d_ws, size_t ws_size,
                              hipStream_t stream)
{
    (void)n_in; (void)ws_size;
    const float*     h     = (const float*)d_in[0];
    const float*     dist  = (const float*)d_in[1];
    const long long* edges = (const long long*)d_in[2];   // int64 [2,E]
    /* d_in[3] node_mask: unused by reference math */
    const float*     emask = (const float*)d_in[4];
    const float*     Wl    = (const float*)d_in[5];
    const float*     bl    = (const float*)d_in[6];
    const float*     W1    = (const float*)d_in[7];
    const float*     b1    = (const float*)d_in[8];
    const float*     w2    = (const float*)d_in[9];
    const float*     b2    = (const float*)d_in[10];
    const float*     lng   = (const float*)d_in[11];
    const float*     lnb   = (const float*)d_in[12];

    int N = in_sizes[0] / DFEAT;   // 100000
    int E = in_sizes[1];           // 1000000

    char* ws = (char*)d_ws;
    float*  x     = (float*)ws;                                   // N*64 f32
    bf16_t* xb    = (bf16_t*)(ws + (size_t)N * DFEAT * 4);        // N*64 bf16
    bf16_t* wfrag = (bf16_t*)(ws + (size_t)N * DFEAT * 6);        // 24*32*16 bf16
    float*  sctab = (float*)(ws + (size_t)N * DFEAT * 6 + 24 * 32 * 16 * 2);
    float*  out   = (float*)d_out;

    // d_out doubles as the scatter accumulator: zero it every call.
    hipMemsetAsync(d_out, 0, (size_t)out_size * sizeof(float), stream);

    // Pre-pack weight B-fragments (bf16, WMMA lane layout) + scalar table.
    pack_kernel<<<1, 256, 0, stream>>>(Wl, W1, b1, w2, wfrag, sctab);

    {   // projection: 1 wave per 16 nodes, 8 waves/block
        int tiles  = N / 16;
        int blocks = (tiles + 7) / 8;
        proj_kernel<<<blocks, 256, 0, stream>>>(h, wfrag, bl, x, xb, N);
    }
    {   // edge MLP + scatter: 1 wave per 16 edges
        int tiles  = E / 16;
        int blocks = (tiles + 7) / 8;
        edge_kernel<<<blocks, 256, 0, stream>>>(xb, x, edges, dist, emask,
                                                wfrag, sctab, b2, out, E);
    }
    {   // residual + LayerNorm + silu, in place
        int blocks = (N + 7) / 8;
        finalize_kernel<<<blocks, 256, 0, stream>>>(out, x, lng, lnb, N);
    }
}
